// MultiDimensionalRecurrentNetwork_66571993088873
// MI455X (gfx1250) — compile-verified, matrix operand order checked by reading, not run on hardware
//
#include <hip/hip_runtime.h>
#include <hip/hip_bf16.h>
#include <math.h>

// Problem constants (from reference): B, Hd, Wd, Cd, D, HID = 64,16,16,32,64,256
#define BB   64
#define HD   16
#define WD   16
#define CD   32
#define DD   64
#define HID  256
#define K3   (3 * HID)        // 768  (W1;W2;W3 concatenated K)
#define KTOT (K3 + DD)        // 832  (+ input projection K)
#define KSTEPS (KTOT / 32)    // 26 WMMA K-steps of 32
#define CELL_ELEMS (BB * HID) // bf16 elems per cell in the hidden mirror

typedef __bf16 bf16_t;
typedef bf16_t v16bf __attribute__((ext_vector_type(16)));
typedef bf16_t v8bf  __attribute__((ext_vector_type(8)));
typedef float  v8f   __attribute__((ext_vector_type(8)));
typedef int    v4i_t __attribute__((ext_vector_type(4)));

// gfx1250 async global->LDS path, guarded so either toolchain compiles.
#if defined(__HIP_DEVICE_COMPILE__) && __has_builtin(__builtin_amdgcn_global_load_async_to_lds_b128)
#define ASYNC_LDS 1
#else
#define ASYNC_LDS 0
#endif

__device__ __forceinline__ float fast_tanh(float xv) {
    // Branchless: tanh(x) = sign(x) * (1 - 2/(e^{2|x|}+1)); e->inf saturates to 1.
    float a = __builtin_fabsf(xv);
    float e = __expf(a + a);                       // native v_exp_f32 path
    float t = 1.0f - 2.0f * __builtin_amdgcn_rcpf(e + 1.0f);
    return __builtin_copysignf(t, xv);
}

__device__ __forceinline__ v8bf pack8(const float4 f0, const float4 f1) {
    v8bf h;
    h[0] = (bf16_t)f0.x; h[1] = (bf16_t)f0.y; h[2] = (bf16_t)f0.z; h[3] = (bf16_t)f0.w;
    h[4] = (bf16_t)f1.x; h[5] = (bf16_t)f1.y; h[6] = (bf16_t)f1.z; h[7] = (bf16_t)f1.w;
    return h;
}

// ---------------------------------------------------------------------------
// Prep: bf16 transposed combined weights Wt[n][k] (HID x KTOT) + fused bias.
// k<768 -> W_h[k][n] ; k>=768 -> W_in[k-768][n].  bias = b_in + b_h.
// ---------------------------------------------------------------------------
__global__ void mdrnn_prep_kernel(const float* __restrict__ W_in,
                                  const float* __restrict__ b_in,
                                  const float* __restrict__ W_h,
                                  const float* __restrict__ b_h,
                                  unsigned short* __restrict__ Wt_u,
                                  float* __restrict__ bias) {
    int idx = blockIdx.x * blockDim.x + threadIdx.x;
    bf16_t* Wt = (bf16_t*)Wt_u;
    if (idx < HID * KTOT) {
        int n = idx / KTOT;
        int k = idx % KTOT;
        float v = (k < K3) ? W_h[(size_t)k * HID + n]
                           : W_in[(size_t)(k - K3) * HID + n];
        Wt[(size_t)n * KTOT + k] = (bf16_t)v;
    }
    if (idx < HID) bias[idx] = b_in[idx] + b_h[idx];
}

// ---------------------------------------------------------------------------
// One workgroup = one grid cell on diagonal `diag`.
// hid = tanh([h_up|h_left|h_back|x] (64x832) @ Wt^T + bias), bf16 WMMA.
// USE_HBUF: hidden state mirrored as bf16 (cell-major) in `hbuf`; A-staging of
// the recurrent K-section becomes a raw copy -> async global->LDS b128.
// ---------------------------------------------------------------------------
template <bool USE_HBUF>
__global__ __launch_bounds__(256, 1)
void mdrnn_cell_kernel(const float* __restrict__ x,             // (B,Hd,Wd,Cd,D)
                       const unsigned short* __restrict__ Wt_u, // (HID,KTOT) bf16
                       const float* __restrict__ bias,          // (HID)
                       float* __restrict__ out,                 // (B,Hd,Wd,Cd,HID)
                       unsigned short* __restrict__ hbuf,       // cell-major bf16 mirror
                       int diag) {
    const bf16_t* __restrict__ Wt = (const bf16_t*)Wt_u;

    // Staged A matrix: 64 rows (batch) x 832 K, bf16.  104 KB of LDS (<320KB).
    __shared__ __align__(16) bf16_t Abuf[BB * KTOT];

    // ---- map blockIdx.x -> (ci,cj,ck) on this anti-diagonal ----------------
    int want = blockIdx.x;
    int ci = 0, cj = 0, ck = 0;
    {
        int cnt = 0;
        bool found = false;
        for (int ii = 0; ii < HD && !found; ++ii)
            for (int jj = 0; jj < WD && !found; ++jj) {
                int kc = diag - ii - jj;
                if (kc >= 0 && kc < CD) {
                    if (cnt == want) { ci = ii; cj = jj; ck = kc; found = true; }
                    ++cnt;
                }
            }
    }

    const int tid = threadIdx.x;
    bool used_async = false;

    if (USE_HBUF) {
        // ---- A-staging, recurrent section: raw bf16 copy from cell-major ---
        // mirror. 3 segments x 32KB; 2048 16-byte chunks each; 8 per thread.
        const unsigned short* seg_src[3];
        seg_src[0] = (ci > 0) ? hbuf + (size_t)(((ci - 1) * WD + cj) * CD + ck) * CELL_ELEMS : nullptr;
        seg_src[1] = (cj > 0) ? hbuf + (size_t)((ci * WD + (cj - 1)) * CD + ck) * CELL_ELEMS : nullptr;
        seg_src[2] = (ck > 0) ? hbuf + (size_t)((ci * WD + cj) * CD + (ck - 1)) * CELL_ELEMS : nullptr;
        #pragma unroll
        for (int seg = 0; seg < 3; ++seg) {
            const unsigned short* src = seg_src[seg];
            if (src) {
                #pragma unroll
                for (int u = 0; u < 8; ++u) {
                    int c = tid + 256 * u;        // 16B chunk id, 0..2047
                    int m = c >> 5;               // 32 chunks per 512B row
                    int o = (c & 31) * 16;        // byte offset within row
                    const char* g = (const char*)src + m * 512 + o;
                    char*       l = (char*)Abuf + m * (KTOT * 2) + seg * 512 + o;
#if ASYNC_LDS
                    __builtin_amdgcn_global_load_async_to_lds_b128(
                        (__attribute__((address_space(1))) v4i_t*)(void*)g,
                        (__attribute__((address_space(3))) v4i_t*)(void*)l, 0, 0);
#else
                    *(v8bf*)l = *(const v8bf*)g;
#endif
                }
                used_async = true;
            } else {
                v8bf z;
                #pragma unroll
                for (int i = 0; i < 8; ++i) z[i] = (bf16_t)0.0f;
                #pragma unroll
                for (int u = 0; u < 8; ++u) {
                    int c = tid + 256 * u;
                    int m = c >> 5;
                    int o = (c & 31) * 16;
                    *(v8bf*)((char*)Abuf + m * (KTOT * 2) + seg * 512 + o) = z;
                }
            }
        }
    } else {
        // ---- A-staging from fp32 hidden state (in `out`), vectorized -------
        const int m  = tid >> 2;        // batch row
        const int cb = (tid & 3) * 64;  // 64 columns per thread per segment
        const float* seg_src[3];
        seg_src[0] = (ci > 0) ? out + ((((size_t)m * HD + (ci - 1)) * WD + cj) * CD + ck) * HID : nullptr;
        seg_src[1] = (cj > 0) ? out + ((((size_t)m * HD + ci) * WD + (cj - 1)) * CD + ck) * HID : nullptr;
        seg_src[2] = (ck > 0) ? out + ((((size_t)m * HD + ci) * WD + cj) * CD + (ck - 1)) * HID : nullptr;
        #pragma unroll
        for (int seg = 0; seg < 3; ++seg) {
            bf16_t* drow = &Abuf[m * KTOT + seg * 256 + cb];
            const float* src = seg_src[seg];
            if (src) {
                #pragma unroll
                for (int c = 0; c < 64; c += 8) {
                    float4 f0 = *(const float4*)(src + cb + c);
                    float4 f1 = *(const float4*)(src + cb + c + 4);
                    *(v8bf*)(drow + c) = pack8(f0, f1);
                }
            } else {
                v8bf z;
                #pragma unroll
                for (int i = 0; i < 8; ++i) z[i] = (bf16_t)0.0f;
                #pragma unroll
                for (int c = 0; c < 64; c += 8) *(v8bf*)(drow + c) = z;
            }
        }
    }

    // ---- A-staging, x-projection section (fp32 -> bf16, 64 cols) -----------
    {
        const int m  = tid >> 2;
        const int cb = (tid & 3) * 16;
        const float* xr = x + ((((size_t)m * HD + ci) * WD + cj) * CD + ck) * DD;
        bf16_t* drow = &Abuf[m * KTOT + K3 + cb];
        #pragma unroll
        for (int c = 0; c < 16; c += 8) {
            float4 f0 = *(const float4*)(xr + cb + c);
            float4 f1 = *(const float4*)(xr + cb + c + 4);
            *(v8bf*)(drow + c) = pack8(f0, f1);
        }
    }

#if ASYNC_LDS
    if (USE_HBUF && used_async) {
#if __has_builtin(__builtin_amdgcn_s_wait_asynccnt)
        __builtin_amdgcn_s_wait_asynccnt(0);
#else
        asm volatile("s_wait_asynccnt 0" ::: "memory");
#endif
    }
#else
    (void)used_async;
#endif
    __syncthreads();

    // ---- WMMA compute ------------------------------------------------------
    // 8 waves; wave w owns N-tiles {2w, 2w+1}, all 4 M-tiles.
    const int w    = tid >> 5;
    const int lane = tid & 31;
    const int lm   = lane & 15;
    const int lh   = lane >> 4;

    v8f acc[4][2];
    #pragma unroll
    for (int nt = 0; nt < 2; ++nt) {
        float bv = bias[(2 * w + nt) * 16 + lm];
        #pragma unroll
        for (int mt = 0; mt < 4; ++mt)
            #pragma unroll
            for (int v = 0; v < 8; ++v)
                acc[mt][nt][v] = bv;
    }

    for (int ks = 0; ks < KSTEPS; ++ks) {
        const int kb = ks * 32 + lh * 8;

        v16bf bfrag[2];
        #pragma unroll
        for (int nt = 0; nt < 2; ++nt) {
            const v8bf* q0 = (const v8bf*)(Wt + (size_t)((2 * w + nt) * 16 + lm) * KTOT + kb);
            bfrag[nt] = __builtin_shufflevector(q0[0], q0[2],
                0, 1, 2, 3, 4, 5, 6, 7, 8, 9, 10, 11, 12, 13, 14, 15);
        }

        #pragma unroll
        for (int mt = 0; mt < 4; ++mt) {
            const v8bf* p0 = (const v8bf*)(&Abuf[(mt * 16 + lm) * KTOT + kb]);
            v16bf a = __builtin_shufflevector(p0[0], p0[2],
                0, 1, 2, 3, 4, 5, 6, 7, 8, 9, 10, 11, 12, 13, 14, 15);
            #pragma unroll
            for (int nt = 0; nt < 2; ++nt) {
                acc[mt][nt] = __builtin_amdgcn_wmma_f32_16x16x32_bf16(
                    false, a, false, bfrag[nt], (short)0, acc[mt][nt], false, false);
            }
        }
    }

    // ---- epilogue: branchless tanh, fp32 out + bf16 mirror -----------------
    bf16_t* hcell = USE_HBUF
        ? (bf16_t*)(hbuf + (size_t)((ci * WD + cj) * CD + ck) * CELL_ELEMS)
        : nullptr;
    #pragma unroll
    for (int mt = 0; mt < 4; ++mt) {
        #pragma unroll
        for (int nt = 0; nt < 2; ++nt) {
            const int n = (2 * w + nt) * 16 + lm;
            #pragma unroll
            for (int v = 0; v < 8; ++v) {
                const int m = mt * 16 + v + lh * 8;   // C/D VGPR layout
                float t = fast_tanh(acc[mt][nt][v]);
                size_t o = ((((size_t)m * HD + ci) * WD + cj) * CD + ck) * (size_t)HID + n;
                out[o] = t;
                if (USE_HBUF) hcell[m * HID + n] = (bf16_t)t;
            }
        }
    }
}

// ---------------------------------------------------------------------------
extern "C" void kernel_launch(void* const* d_in, const int* in_sizes, int n_in,
                              void* d_out, int out_size, void* d_ws, size_t ws_size,
                              hipStream_t stream) {
    const float* x    = (const float*)d_in[0];
    const float* W_in = (const float*)d_in[1];
    const float* b_in = (const float*)d_in[2];
    const float* W_h  = (const float*)d_in[3];
    const float* b_h  = (const float*)d_in[4];
    float* out = (float*)d_out;

    // Workspace: [Wt bf16 | bias fp32] at 0, bf16 hidden mirror at +512KB.
    unsigned short* Wt = (unsigned short*)d_ws;
    float* bias = (float*)((char*)d_ws + (size_t)HID * KTOT * sizeof(unsigned short));
    const size_t hbuf_off  = (size_t)512 * 1024;
    const size_t hbuf_size = (size_t)HD * WD * CD * CELL_ELEMS * sizeof(unsigned short); // 256 MB
    unsigned short* hbuf = (unsigned short*)((char*)d_ws + hbuf_off);
    const bool use_hbuf = ws_size >= hbuf_off + hbuf_size;

    {
        int total = HID * KTOT;
        mdrnn_prep_kernel<<<(total + 255) / 256, 256, 0, stream>>>(
            W_in, b_in, W_h, b_h, Wt, bias);
    }

    // 3-D wavefront: one launch per anti-diagonal d = i + j + k.
    for (int d = 0; d < HD + WD + CD - 2; ++d) {
        int cnt = 0;
        for (int i = 0; i < HD; ++i)
            for (int j = 0; j < WD; ++j) {
                int k = d - i - j;
                if (k >= 0 && k < CD) ++cnt;
            }
        if (cnt <= 0) continue;
        if (use_hbuf)
            mdrnn_cell_kernel<true><<<cnt, 256, 0, stream>>>(x, Wt, bias, out, hbuf, d);
        else
            mdrnn_cell_kernel<false><<<cnt, 256, 0, stream>>>(x, Wt, bias, out, hbuf, d);
    }
}
